// BasicTransformerBlock_25821343383557
// MI455X (gfx1250) — matrix operand order repaired and checked by reading, not compiled, not run on hardware
//
#include <hip/hip_runtime.h>
#include <hip/hip_bf16.h>
#include <math.h>

typedef __attribute__((ext_vector_type(16))) _Float16 v16h;
typedef __attribute__((ext_vector_type(8)))  _Float16 v8h;
typedef __attribute__((ext_vector_type(4)))  _Float16 v4h;
typedef __attribute__((ext_vector_type(8)))  float    v8f;

#define NHEADS 16
#define DHEAD  64

__device__ inline v16h pack16(v8h lo, v8h hi) {
    v16h r;
#pragma unroll
    for (int i = 0; i < 8; ++i) { r[i] = lo[i]; r[i + 8] = hi[i]; }
    return r;
}

__device__ inline v8f wmma_f16(v16h a, v16h b, v8f c) {
    // D = A(16x32 f16) x B(32x16 f16) + C(16x16 f32)
    return __builtin_amdgcn_wmma_f32_16x16x32_f16(
        false, a, false, b, (short)0, c, false, false);
}

// ---------------------------------------------------------------------------
// fp32 -> f16 conversion (n multiple of 4)
// ---------------------------------------------------------------------------
__global__ __launch_bounds__(256) void cvt_f16_kernel(
    const float* __restrict__ in, _Float16* __restrict__ out, long n4) {
    long i = (long)blockIdx.x * blockDim.x + threadIdx.x;
    if (i < n4) {
        float4 v = ((const float4*)in)[i];
        v4h o;
        o[0] = (_Float16)v.x; o[1] = (_Float16)v.y;
        o[2] = (_Float16)v.z; o[3] = (_Float16)v.w;
        ((v4h*)out)[i] = o;
    }
}

// ---------------------------------------------------------------------------
// fp32 [Kr][Nc] -> f16 transposed [Nc][Kr]  (both dims multiples of 32)
// ---------------------------------------------------------------------------
__global__ __launch_bounds__(256) void cvt_tr_kernel(
    const float* __restrict__ in, _Float16* __restrict__ out, int Kr, int Nc) {
    __shared__ float tile[32][33];
    int n0 = blockIdx.x * 32, k0 = blockIdx.y * 32;
    int tx = threadIdx.x, ty = threadIdx.y;   // 32 x 8
#pragma unroll
    for (int i = 0; i < 4; ++i)
        tile[ty + i * 8][tx] = in[(size_t)(k0 + ty + i * 8) * Nc + n0 + tx];
    __syncthreads();
#pragma unroll
    for (int i = 0; i < 4; ++i)
        out[(size_t)(n0 + ty + i * 8) * Kr + k0 + tx] = (_Float16)tile[tx][ty + i * 8];
}

// ---------------------------------------------------------------------------
// LayerNorm over C=1024, fp32 in, f16 out. One block (256 thr) per row.
// ---------------------------------------------------------------------------
__global__ __launch_bounds__(256) void ln_f16_kernel(
    const float* __restrict__ x, const float* __restrict__ gamma,
    const float* __restrict__ beta, _Float16* __restrict__ out) {
    const int C = 1024;
    const float* xr = x + (size_t)blockIdx.x * C;
    int t = threadIdx.x;
    float4 v = ((const float4*)xr)[t];
    float s = v.x + v.y + v.z + v.w;
    float q = v.x * v.x + v.y * v.y + v.z * v.z + v.w * v.w;
#pragma unroll
    for (int off = 16; off > 0; off >>= 1) {
        s += __shfl_xor(s, off, 32);
        q += __shfl_xor(q, off, 32);
    }
    __shared__ float shs[8], shq[8];
    int wid = t >> 5, lane = t & 31;
    if (lane == 0) { shs[wid] = s; shq[wid] = q; }
    __syncthreads();
    if (t == 0) {
        float ts = 0.f, tq = 0.f;
#pragma unroll
        for (int i = 0; i < 8; ++i) { ts += shs[i]; tq += shq[i]; }
        shs[0] = ts; shq[0] = tq;
    }
    __syncthreads();
    float mean = shs[0] * (1.0f / C);
    float var  = shq[0] * (1.0f / C) - mean * mean;
    float inv  = rsqrtf(var + 1e-5f);
    float4 g = ((const float4*)gamma)[t];
    float4 b = ((const float4*)beta)[t];
    v4h o;
    o[0] = (_Float16)((v.x - mean) * inv * g.x + b.x);
    o[1] = (_Float16)((v.y - mean) * inv * g.y + b.y);
    o[2] = (_Float16)((v.z - mean) * inv * g.z + b.z);
    o[3] = (_Float16)((v.w - mean) * inv * g.w + b.w);
    ((v4h*)(out + (size_t)blockIdx.x * C))[t] = o;
}

// ---------------------------------------------------------------------------
// WMMA GEMM: C[M,Nc] = A[M,K] @ Bt[Nc,K]^T   (f16 in, f32 accumulate)
// MODE 0: f16 out
// MODE 1: f32 out = acc + bias[col] + res[row,col]
// MODE 2: GEGLU (gate weights at Bt rows [Nc,2*Nc))
// Block tile 128x128, 8 waves (4 M x 2 N); double-buffered LDS.
// ---------------------------------------------------------------------------
template <int MODE>
__global__ __launch_bounds__(256) void gemm_kernel(
    const _Float16* __restrict__ A, const _Float16* __restrict__ Bt,
    _Float16* __restrict__ Ch, float* __restrict__ Cf,
    const float* __restrict__ bias, const float* __restrict__ res,
    int M, int Nc, int K) {
    __shared__ alignas(16) _Float16 As[2][128][40];
    __shared__ alignas(16) _Float16 Bs[(MODE == 2) ? 4 : 2][128][40];

    int t = threadIdx.x;
    int wid = t >> 5, lane = t & 31;
    int l16 = lane & 15, h16 = (lane >> 4) & 1;
    int wm = wid & 3, wn = wid >> 2;
    int blockM = blockIdx.y * 128, blockN = blockIdx.x * 128;

    v8f zacc = {};
    v8h zh = {};
    v8f acc[2][4], accg[2][4];
#pragma unroll
    for (int mi = 0; mi < 2; ++mi)
#pragma unroll
        for (int ni = 0; ni < 4; ++ni) { acc[mi][ni] = zacc; accg[mi][ni] = zacc; }

    int ra = t >> 1, ca = (t & 1) << 4;   // 128 rows x 32 k per stage
    int growA = blockM + ra;
    int grownB = blockN + ra;

    v8h a0, a1, b0, b1, g0, g1;
    auto loadT = [&](int k0) {
        a0 = zh; a1 = zh;
        if (growA < M) {
            const v8h* ap = (const v8h*)(A + (size_t)growA * K + k0 + ca);
            a0 = ap[0]; a1 = ap[1];
        }
        const v8h* bp = (const v8h*)(Bt + (size_t)grownB * K + k0 + ca);
        b0 = bp[0]; b1 = bp[1];
        if constexpr (MODE == 2) {
            const v8h* gp = (const v8h*)(Bt + (size_t)(grownB + Nc) * K + k0 + ca);
            g0 = gp[0]; g1 = gp[1];
        }
    };
    auto storeT = [&](int buf) {
        *(v8h*)&As[buf][ra][ca]     = a0;
        *(v8h*)&As[buf][ra][ca + 8] = a1;
        *(v8h*)&Bs[buf][ra][ca]     = b0;
        *(v8h*)&Bs[buf][ra][ca + 8] = b1;
        if constexpr (MODE == 2) {
            *(v8h*)&Bs[2 + buf][ra][ca]     = g0;
            *(v8h*)&Bs[2 + buf][ra][ca + 8] = g1;
        }
    };

    int nk = K >> 5;
    loadT(0);
    storeT(0);
    __syncthreads();

    for (int kt = 0; kt < nk; ++kt) {
        int cur = kt & 1;
        bool more = (kt + 1) < nk;
        if (more) loadT((kt + 1) << 5);   // overlap next tile's VMEM with WMMA

        v16h af[2], bf[4], gf[4];
#pragma unroll
        for (int mi = 0; mi < 2; ++mi) {
            int ml = wm * 32 + mi * 16 + l16;
            af[mi] = pack16(*(const v8h*)&As[cur][ml][h16 * 8],
                            *(const v8h*)&As[cur][ml][16 + h16 * 8]);
        }
#pragma unroll
        for (int ni = 0; ni < 4; ++ni) {
            int nl = wn * 64 + ni * 16 + l16;
            bf[ni] = pack16(*(const v8h*)&Bs[cur][nl][h16 * 8],
                            *(const v8h*)&Bs[cur][nl][16 + h16 * 8]);
            if constexpr (MODE == 2)
                gf[ni] = pack16(*(const v8h*)&Bs[2 + cur][nl][h16 * 8],
                                *(const v8h*)&Bs[2 + cur][nl][16 + h16 * 8]);
        }
#pragma unroll
        for (int mi = 0; mi < 2; ++mi)
#pragma unroll
            for (int ni = 0; ni < 4; ++ni) {
                acc[mi][ni] = wmma_f16(af[mi], bf[ni], acc[mi][ni]);
                if constexpr (MODE == 2)
                    accg[mi][ni] = wmma_f16(af[mi], gf[ni], accg[mi][ni]);
            }
        if (more) storeT(cur ^ 1);
        __syncthreads();
    }

#pragma unroll
    for (int mi = 0; mi < 2; ++mi)
#pragma unroll
        for (int ni = 0; ni < 4; ++ni)
#pragma unroll
            for (int r = 0; r < 8; ++r) {
                int row = blockM + wm * 32 + mi * 16 + r + h16 * 8;
                int col = blockN + wn * 64 + ni * 16 + l16;
                if (row < M) {
                    float v = acc[mi][ni][r];
                    if constexpr (MODE == 0) {
                        Ch[(size_t)row * Nc + col] = (_Float16)v;
                    } else if constexpr (MODE == 1) {
                        Cf[(size_t)row * Nc + col] =
                            v + bias[col] + res[(size_t)row * Nc + col];
                    } else {
                        float a = v + bias[col];
                        float g = accg[mi][ni][r] + bias[col + Nc];
                        float ge = 0.5f * g * (1.0f + erff(g * 0.70710678118654752f));
                        Ch[(size_t)row * Nc + col] = (_Float16)(a * ge);
                    }
                }
            }
}

// ---------------------------------------------------------------------------
// Flash attention. q/o: [B*N, 1024] f16, k/v: [B*S, 1024] f16 (heads packed).
// Grid: (N/128, B*NHEADS). Each wave owns 16 q rows; key tiles of 64.
// Full tiles (always, for self-attn) take a uniform unguarded path: K
// fragments batch-load into registers, then 8 WMMAs; no exec-mask churn.
// Row sums come free via a ones-column WMMA accumulator. V double-buffered.
// ---------------------------------------------------------------------------
__global__ __launch_bounds__(256) void attn_kernel(
    const _Float16* __restrict__ Q, const _Float16* __restrict__ Km,
    const _Float16* __restrict__ Vm, _Float16* __restrict__ O,
    int N, int S) {
    __shared__ alignas(16) _Float16 Vt[2][64][72];     // V^T tiles: [d][key]
    __shared__ alignas(16) _Float16 Pst[8][16][72];    // per-wave P staging

    int b = blockIdx.y >> 4;
    int h = blockIdx.y & 15;
    int t = threadIdx.x;
    int wid = t >> 5, lane = t & 31;
    int l16 = lane & 15, h16 = (lane >> 4) & 1;
    int qrow0 = blockIdx.x * 128 + wid * 16;

    const _Float16* Kb = Km + (size_t)b * S * 1024 + h * DHEAD;   // head base
    const _Float16* Vb = Vm + (size_t)b * S * 1024 + h * DHEAD;

    // Q fragments (A-layout, straight from global)
    const _Float16* qp = Q + (size_t)(b * N + qrow0 + l16) * 1024 + h * DHEAD;
    v16h qa[2];
#pragma unroll
    for (int c = 0; c < 2; ++c)
        qa[c] = pack16(*(const v8h*)(qp + c * 32 + h16 * 8),
                       *(const v8h*)(qp + c * 32 + 16 + h16 * 8));

    v8f zacc = {};
    v8h zh = {};
    v16h ones;
#pragma unroll
    for (int i = 0; i < 16; ++i) ones[i] = (_Float16)1.0f;

    float m[8];
    v8f oacc[4], lacc = zacc;
#pragma unroll
    for (int r = 0; r < 8; ++r) m[r] = -3.0e38f;
#pragma unroll
    for (int d = 0; d < 4; ++d) oacc[d] = zacc;

    // V staging: thread t handles key (t>>2), d-range [(t&3)*16, +16)
    int keyl = t >> 2;
    int d0 = (t & 3) << 4;
    v8h v0, v1;
    auto loadV = [&](int kt2) {
        int key = kt2 * 64 + keyl;
        if (kt2 * 64 + 64 <= S) {                       // uniform: full tile
            const _Float16* vp = Vb + (size_t)key * 1024 + d0;
            v0 = *(const v8h*)vp; v1 = *(const v8h*)(vp + 8);
        } else {
            v0 = zh; v1 = zh;
            if (key < S) {
                const _Float16* vp = Vb + (size_t)key * 1024 + d0;
                v0 = *(const v8h*)vp; v1 = *(const v8h*)(vp + 8);
            }
        }
    };
    auto storeV = [&](int buf) {
#pragma unroll
        for (int i = 0; i < 8; ++i) {
            Vt[buf][d0 + i][keyl]     = v0[i];
            Vt[buf][d0 + 8 + i][keyl] = v1[i];
        }
    };

    int ktiles = (S + 63) >> 6;
    loadV(0);
    storeV(0);
    __syncthreads();

    for (int kt = 0; kt < ktiles; ++kt) {
        int cur = kt & 1;
        bool more = (kt + 1) < ktiles;
        bool full = (kt * 64 + 64) <= S;   // uniform across the block
        if (more) loadV(kt + 1);           // overlap next V tile's VMEM

        // ---- K fragments: batch-load all 8, then batch WMMA ----
        v16h kf[4][2];
        if (full) {
#pragma unroll
            for (int nt = 0; nt < 4; ++nt) {
                const _Float16* kp =
                    Kb + (size_t)(kt * 64 + nt * 16 + l16) * 1024 + h16 * 8;
#pragma unroll
                for (int c = 0; c < 2; ++c)
                    kf[nt][c] = pack16(*(const v8h*)(kp + c * 32),
                                       *(const v8h*)(kp + c * 32 + 16));
            }
        } else {
#pragma unroll
            for (int nt = 0; nt < 4; ++nt) {
                int keyg = kt * 64 + nt * 16 + l16;
#pragma unroll
                for (int c = 0; c < 2; ++c) kf[nt][c] = pack16(zh, zh);
                if (keyg < S) {
                    const _Float16* kp = Kb + (size_t)keyg * 1024 + h16 * 8;
#pragma unroll
                    for (int c = 0; c < 2; ++c)
                        kf[nt][c] = pack16(*(const v8h*)(kp + c * 32),
                                           *(const v8h*)(kp + c * 32 + 16));
                }
            }
        }
        v8f s[4];
#pragma unroll
        for (int nt = 0; nt < 4; ++nt) s[nt] = zacc;
#pragma unroll
        for (int nt = 0; nt < 4; ++nt)
#pragma unroll
            for (int c = 0; c < 2; ++c)
                s[nt] = wmma_f16(qa[c], kf[nt][c], s[nt]);

        // ---- scale (+ mask only on ragged tile) ----
        if (full) {
#pragma unroll
            for (int nt = 0; nt < 4; ++nt)
#pragma unroll
                for (int r = 0; r < 8; ++r)
                    s[nt][r] *= 0.125f;
        } else {
#pragma unroll
            for (int nt = 0; nt < 4; ++nt) {
                bool valid = (kt * 64 + nt * 16 + l16) < S;
#pragma unroll
                for (int r = 0; r < 8; ++r)
                    s[nt][r] = valid ? s[nt][r] * 0.125f : -3.0e38f;
            }
        }
        // ---- online max (row reduction within 16-lane halves) ----
        float alpha[8];
#pragma unroll
        for (int r = 0; r < 8; ++r) {
            float mv = fmaxf(fmaxf(s[0][r], s[1][r]), fmaxf(s[2][r], s[3][r]));
#pragma unroll
            for (int off = 1; off < 16; off <<= 1)
                mv = fmaxf(mv, __shfl_xor(mv, off, 32));
            float mn = fmaxf(m[r], mv);
            alpha[r] = __expf(m[r] - mn);
            m[r] = mn;
        }
#pragma unroll
        for (int nt = 0; nt < 4; ++nt)
#pragma unroll
            for (int r = 0; r < 8; ++r)
                s[nt][r] = __expf(s[nt][r] - m[r]);
        // ---- rescale running O and l ----
#pragma unroll
        for (int r = 0; r < 8; ++r) {
            lacc[r] *= alpha[r];
#pragma unroll
            for (int d = 0; d < 4; ++d) oacc[d][r] *= alpha[r];
        }
        // ---- P: C-layout -> f16 -> per-wave LDS -> A-layout fragments ----
#pragma unroll
        for (int nt = 0; nt < 4; ++nt)
#pragma unroll
            for (int r = 0; r < 8; ++r)
                Pst[wid][r + h16 * 8][nt * 16 + l16] = (_Float16)s[nt][r];
        v16h pa[2];
#pragma unroll
        for (int c = 0; c < 2; ++c) {
            const _Float16* pb = &Pst[wid][l16][c * 32];
            pa[c] = pack16(*(const v8h*)(pb + h16 * 8),
                           *(const v8h*)(pb + 16 + h16 * 8));
        }
        // ---- O += P @ V ; l += P @ 1 (row sums replicated across lanes) ----
#pragma unroll
        for (int c = 0; c < 2; ++c) {
#pragma unroll
            for (int dt = 0; dt < 4; ++dt) {
                const _Float16* vb = &Vt[cur][dt * 16 + l16][c * 32 + h16 * 8];
                v16h vf = pack16(*(const v8h*)vb, *(const v8h*)(vb + 16));
                oacc[dt] = wmma_f16(pa[c], vf, oacc[dt]);
            }
            lacc = wmma_f16(pa[c], ones, lacc);
        }
        if (more) storeV(cur ^ 1);
        __syncthreads();
    }
    // ---- normalize + store ----
#pragma unroll
    for (int dt = 0; dt < 4; ++dt)
#pragma unroll
        for (int r = 0; r < 8; ++r) {
            int row = qrow0 + r + h16 * 8;
            float ov = oacc[dt][r] / lacc[r];
            O[(size_t)(b * N + row) * 1024 + h * DHEAD + dt * 16 + l16] = (_Float16)ov;
        }
}

// ---------------------------------------------------------------------------
// Host orchestration
// ---------------------------------------------------------------------------
extern "C" void kernel_launch(void* const* d_in, const int* in_sizes, int n_in,
                              void* d_out, int out_size, void* d_ws, size_t ws_size,
                              hipStream_t stream) {
    const int B = 4, N = 2048, S = 77, DIM = 1024, CTX = 768, FF = 4096;
    const int rows  = B * N;   // 8192
    const int crows = B * S;   // 308

    const float* x    = (const float*)d_in[0];
    const float* ctx  = (const float*)d_in[1];
    const float* Wq1  = (const float*)d_in[2];
    const float* Wk1  = (const float*)d_in[3];
    const float* Wv1  = (const float*)d_in[4];
    const float* Wo1  = (const float*)d_in[5];
    const float* bo1  = (const float*)d_in[6];
    const float* Wq2  = (const float*)d_in[7];
    const float* Wk2  = (const float*)d_in[8];
    const float* Wv2  = (const float*)d_in[9];
    const float* Wo2  = (const float*)d_in[10];
    const float* bo2  = (const float*)d_in[11];
    const float* Wp   = (const float*)d_in[12];
    const float* bp   = (const float*)d_in[13];
    const float* Wf   = (const float*)d_in[14];
    const float* bf   = (const float*)d_in[15];
    const float* g1   = (const float*)d_in[16];
    const float* b1   = (const float*)d_in[17];
    const float* g2   = (const float*)d_in[18];
    const float* b2   = (const float*)d_in[19];
    const float* g3   = (const float*)d_in[20];
    const float* b3   = (const float*)d_in[21];
    float* out = (float*)d_out;

    // workspace carving (256B aligned)
    char* p = (char*)d_ws;
    auto a16 = [&](size_t nh) {
        _Float16* r = (_Float16*)p; p += ((nh * 2 + 255) / 256) * 256; return r;
    };
    auto a32 = [&](size_t nf) {
        float* r = (float*)p; p += ((nf * 4 + 255) / 256) * 256; return r;
    };
    _Float16* wq1 = a16((size_t)DIM * DIM);       // all weights stored transposed
    _Float16* wk1 = a16((size_t)DIM * DIM);
    _Float16* wv1 = a16((size_t)DIM * DIM);
    _Float16* wo1 = a16((size_t)DIM * DIM);
    _Float16* wq2 = a16((size_t)DIM * DIM);
    _Float16* wk2 = a16((size_t)CTX * DIM);
    _Float16* wv2 = a16((size_t)CTX * DIM);
    _Float16* wo2 = a16((size_t)DIM * DIM);
    _Float16* wp  = a16((size_t)DIM * 2 * FF);
    _Float16* wf  = a16((size_t)FF * DIM);
    _Float16* lnb = a16((size_t)rows * DIM);
    _Float16* qb  = a16((size_t)rows * DIM);
    _Float16* kb  = a16((size_t)rows * DIM);
    _Float16* vb  = a16((size_t)rows * DIM);
    _Float16* ab  = a16((size_t)rows * DIM);
    _Float16* ch  = a16((size_t)crows * CTX);
    _Float16* ffh = a16((size_t)rows * FF);
    float*    xb  = a32((size_t)rows * DIM);

    // one-time: transpose+convert weights  (Bt[n][k] layouts)
    auto cvtT = [&](const float* src, _Float16* dst, int Kr, int Nc) {
        cvt_tr_kernel<<<dim3(Nc / 32, Kr / 32), dim3(32, 8), 0, stream>>>(src, dst, Kr, Nc);
    };
    cvtT(Wq1, wq1, DIM, DIM);
    cvtT(Wk1, wk1, DIM, DIM);
    cvtT(Wv1, wv1, DIM, DIM);
    cvtT(Wo1, wo1, DIM, DIM);
    cvtT(Wq2, wq2, DIM, DIM);
    cvtT(Wk2, wk2, CTX, DIM);
    cvtT(Wv2, wv2, CTX, DIM);
    cvtT(Wo2, wo2, DIM, DIM);
    cvtT(Wp,  wp,  DIM, 2 * FF);
    cvtT(Wf,  wf,  FF, DIM);
    {   // context stays row-major (A operand of the K/V GEMMs)
        long n4 = (long)((size_t)crows * CTX / 4);
        cvt_f16_kernel<<<dim3((unsigned)((n4 + 255) / 256)), 256, 0, stream>>>(ctx, ch, n4);
    }

    // running residual x (fp32), updated in place
    hipMemcpyAsync(xb, x, (size_t)rows * DIM * sizeof(float),
                   hipMemcpyDeviceToDevice, stream);

    auto gemm0 = [&](const _Float16* A, const _Float16* Bm, _Float16* C,
                     int M, int Nc, int K) {
        gemm_kernel<0><<<dim3(Nc / 128, (M + 127) / 128), 256, 0, stream>>>(
            A, Bm, C, nullptr, nullptr, nullptr, M, Nc, K);
    };
    auto gemm1 = [&](const _Float16* A, const _Float16* Bm, float* C,
                     const float* bias, const float* res, int M, int Nc, int K) {
        gemm_kernel<1><<<dim3(Nc / 128, (M + 127) / 128), 256, 0, stream>>>(
            A, Bm, nullptr, C, bias, res, M, Nc, K);
    };

    // ---- stage 1: self-attention ----
    ln_f16_kernel<<<rows, 256, 0, stream>>>(xb, g1, b1, lnb);
    gemm0(lnb, wq1, qb, rows, DIM, DIM);
    gemm0(lnb, wk1, kb, rows, DIM, DIM);
    gemm0(lnb, wv1, vb, rows, DIM, DIM);
    attn_kernel<<<dim3(N / 128, B * NHEADS), 256, 0, stream>>>(qb, kb, vb, ab, N, N);
    gemm1(ab, wo1, xb, bo1, xb, rows, DIM, DIM);   // x = attn@Wo1 + bo1 + x

    // ---- stage 2: cross-attention ----
    ln_f16_kernel<<<rows, 256, 0, stream>>>(xb, g2, b2, lnb);
    gemm0(lnb, wq2, qb, rows, DIM, DIM);
    gemm0(ch,  wk2, kb, crows, DIM, CTX);
    gemm0(ch,  wv2, vb, crows, DIM, CTX);
    attn_kernel<<<dim3(N / 128, B * NHEADS), 256, 0, stream>>>(qb, kb, vb, ab, N, S);
    gemm1(ab, wo2, xb, bo2, xb, rows, DIM, DIM);   // x = attn@Wo2 + bo2 + x

    // ---- stage 3: GEGLU feed-forward ----
    ln_f16_kernel<<<rows, 256, 0, stream>>>(xb, g3, b3, lnb);
    gemm_kernel<2><<<dim3(FF / 128, rows / 128), 256, 0, stream>>>(
        lnb, wp, ffh, nullptr, bp, nullptr, rows, FF, DIM);
    gemm1(ffh, wf, out, bf, xb, rows, DIM, FF);    // out = geglu@Wf + bf + x
}
